// GraphTransformerNetwork_31799937859996
// MI455X (gfx1250) — matrix operand order repaired and checked by reading, not compiled, not run on hardware
//
#include <hip/hip_runtime.h>
#include <hip/hip_bf16.h>

typedef __attribute__((ext_vector_type(16))) _Float16 v16h;
typedef __attribute__((ext_vector_type(8)))  float    v8f;

#define NN 50000
#define EE 800000
#define ETOT (EE + NN)
#define HH 4
#define CC 16
#define DD 64
#define GG 64
#define NCLS 10
#define NEG_SLOPE 0.2f
#define EPSF 1e-16f
// ordered encoding of -inf: bits(-inf)=0xFF800000 (neg) -> ^0x7FFFFFFF
#define ORD_NEG_INF ((int)0x807FFFFF)

// ---- order-preserving float <-> int for integer atomicMax ----
__device__ __forceinline__ int ordf(float f) {
    int b = __float_as_int(f);
    return (b >= 0) ? b : (b ^ 0x7FFFFFFF);
}
__device__ __forceinline__ float unordf(int b) {
    return __int_as_float((b >= 0) ? b : (b ^ 0x7FFFFFFF));
}

// =====================================================================
// WMMA GEMM:  Out[nRows,64] = X[nRows,64] @ W[64,64] (+ bias[64])
// Block = 256 threads = 8 waves; wave w handles rows [blk*128 + w*16, +16).
// W staged to LDS as f16, column-major (Wl[col][k]) so a B fragment is a
// single contiguous 16-half read per lane.
// Output goes through an LDS tile so global stores are coalesced b128.
// =====================================================================
__global__ void k_gemm64(const float* __restrict__ X, const float* __restrict__ W,
                         const float* __restrict__ bias, float* __restrict__ Out,
                         int nRows) {
    __shared__ __align__(128) _Float16 Wl[64][64];   // [col][k]
    __shared__ __align__(16)  float    Ot[8][16][64]; // per-wave 16x64 out tile
    int tid = threadIdx.x;
    for (int i = tid; i < 64 * 64; i += 256) {
        int k = i >> 6, n = i & 63;                   // W is row-major [k][n]
        Wl[n][k] = (_Float16)W[i];
    }
    __syncthreads();

    int wave = tid >> 5, lane = tid & 31;
    int rowBase = blockIdx.x * 128 + wave * 16;
    int lr   = lane & 15;
    int koff = (lane < 16) ? 0 : 16;                  // A fragment K sub-offset
    int bko  = (lane < 16) ? 0 : 16;                  // B fragment K sub-offset
    int mb   = (lane < 16) ? 0 : 8;                   // C fragment M sub-offset

    // --- A fragments (K blocks 0-31, 32-63), contiguous k = 16*(lane/16) + s order, same as B ---
    int arow = rowBase + lr;
    if (arow >= nRows) arow = nRows - 1;              // clamp read; store guarded
    const float* xr = X + (size_t)arow * 64;
    v16h a0, a1;
#pragma unroll
    for (int j = 0; j < 8; ++j) {
        a0[j]     = (_Float16)xr[koff + j];
        a0[8 + j] = (_Float16)xr[8 + koff + j];
        a1[j]     = (_Float16)xr[32 + koff + j];
        a1[8 + j] = (_Float16)xr[40 + koff + j];
    }

#pragma unroll
    for (int ct = 0; ct < 4; ++ct) {                  // 4 column tiles of 16
        int col = ct * 16 + lr;
        v16h b0 = *(const v16h*)&Wl[col][0  + bko];   // K = 0..31 half
        v16h b1 = *(const v16h*)&Wl[col][32 + bko];   // K = 32..63 half
        v8f c = {};
        c = __builtin_amdgcn_wmma_f32_16x16x32_f16(false, a0, false, b0,
                                                   (short)0, c, false, false);
        c = __builtin_amdgcn_wmma_f32_16x16x32_f16(false, a1, false, b1,
                                                   (short)0, c, false, false);
        float bv = bias ? bias[col] : 0.0f;
#pragma unroll
        for (int r = 0; r < 8; ++r)
            Ot[wave][mb + r][col] = c[r] + bv;        // LDS scatter (fast)
    }
    __syncthreads();                                  // make tiles visible

    // coalesced readback: 2 rows (128B*4) per iteration per wave
    const float4* tp = (const float4*)&Ot[wave][0][0]; // 256 float4's
#pragma unroll
    for (int it = 0; it < 8; ++it) {
        int idx = it * 32 + lane;                     // float4 index in tile
        int row = rowBase + (idx >> 4);
        if (row < nRows) {
            float4 v = tp[idx];
            *(float4*)&Out[(size_t)row * 64 + (idx & 15) * 4] = v;
        }
    }
}

// ---- fills ----
__global__ void k_fill_f32(float* p, float v, int n) {
    int i = blockIdx.x * blockDim.x + threadIdx.x;
    if (i < n) p[i] = v;
}
__global__ void k_fill_i32(int* p, int v, int n) {
    int i = blockIdx.x * blockDim.x + threadIdx.x;
    if (i < n) p[i] = v;
}

// ---- per-node attention logits: als/ald[n,h] = <h[n,h,:], a_src/dst[h,:]> ----
__global__ void k_scores(const float* __restrict__ hbuf,
                         const float* __restrict__ asrc, const float* __restrict__ adst,
                         float* __restrict__ als, float* __restrict__ ald) {
    int idx = blockIdx.x * blockDim.x + threadIdx.x;
    if (idx >= NN * HH) return;
    int n = idx >> 2, hh = idx & 3;
    const float* hr = hbuf + (size_t)n * 64 + hh * 16;
    float s = 0.f, d = 0.f;
#pragma unroll
    for (int c = 0; c < 16; ++c) {
        float hv = hr[c];
        s += hv * asrc[hh * 16 + c];
        d += hv * adst[hh * 16 + c];
    }
    als[idx] = s;
    ald[idx] = d;
}

__device__ __forceinline__ void edge_sd(int i, const int* __restrict__ edge,
                                        int& s, int& d) {
    if (i < EE) { s = edge[i]; d = edge[EE + i]; }
    else        { s = d = i - EE; }                   // self-loops appended
}

// ---- pass 1: leaky-relu score, store, segment max over dst ----
__global__ void k_edge_max(const int* __restrict__ edge,
                           const float* __restrict__ als, const float* __restrict__ ald,
                           float* __restrict__ esc, int* __restrict__ emax) {
    int idx = blockIdx.x * blockDim.x + threadIdx.x;
    if (idx >= ETOT * HH) return;
    int i = idx >> 2, hh = idx & 3;
    int s, d; edge_sd(i, edge, s, d);
    float v = als[s * HH + hh] + ald[d * HH + hh];
    v = (v > 0.f) ? v : NEG_SLOPE * v;
    esc[idx] = v;
    atomicMax(&emax[d * HH + hh], ordf(v));
}

// ---- pass 2: exp-sum denominator ----
__global__ void k_edge_sum(const int* __restrict__ edge,
                           const float* __restrict__ esc, const int* __restrict__ emax,
                           float* __restrict__ denom) {
    int idx = blockIdx.x * blockDim.x + threadIdx.x;
    if (idx >= ETOT * HH) return;
    int i = idx >> 2, hh = idx & 3;
    int s, d; edge_sd(i, edge, s, d);
    float ex = __expf(esc[idx] - unordf(emax[d * HH + hh]));
    atomicAdd(&denom[d * HH + hh], ex);
}

// ---- pass 3: alpha-weighted scatter of features ----
__global__ void k_edge_agg(const int* __restrict__ edge,
                           const float* __restrict__ esc, const int* __restrict__ emax,
                           const float* __restrict__ denom,
                           const float* __restrict__ hbuf, float* __restrict__ obuf) {
    int idx = blockIdx.x * blockDim.x + threadIdx.x;
    if (idx >= ETOT * HH) return;
    int i = idx >> 2, hh = idx & 3;
    int s, d; edge_sd(i, edge, s, d);
    float ex = __expf(esc[idx] - unordf(emax[d * HH + hh]));
    float alpha = ex / (denom[d * HH + hh] + EPSF);
    const float* hs = hbuf + (size_t)s * 64 + hh * 16;
    float* od = obuf + (size_t)d * 64 + hh * 16;
#pragma unroll
    for (int c = 0; c < 16; ++c) atomicAdd(&od[c], alpha * hs[c]);
}

// ---- bias + relu, in place ----
__global__ void k_bias_relu(float* __restrict__ buf, const float* __restrict__ b) {
    int idx = blockIdx.x * blockDim.x + threadIdx.x;
    if (idx >= NN * 64) return;
    float v = buf[idx] + b[idx & 63];
    buf[idx] = (v > 0.f) ? v : 0.f;
}

// ---- global attention pooling (per-graph, per-feature softmax gate) ----
__global__ void k_pool_max(const float* __restrict__ gate, const int* __restrict__ batch,
                           int* __restrict__ gmax) {
    int idx = blockIdx.x * blockDim.x + threadIdx.x;
    if (idx >= NN * 64) return;
    int n = idx >> 6, d = idx & 63;
    atomicMax(&gmax[batch[n] * 64 + d], ordf(gate[idx]));
}
__global__ void k_pool_sum(const float* __restrict__ gate, const int* __restrict__ batch,
                           const int* __restrict__ gmax, float* __restrict__ gden) {
    int idx = blockIdx.x * blockDim.x + threadIdx.x;
    if (idx >= NN * 64) return;
    int n = idx >> 6, d = idx & 63;
    int g = batch[n] * 64 + d;
    atomicAdd(&gden[g], __expf(gate[idx] - unordf(gmax[g])));
}
__global__ void k_pool_acc(const float* __restrict__ gate, const float* __restrict__ xf,
                           const int* __restrict__ batch, const int* __restrict__ gmax,
                           const float* __restrict__ gden, float* __restrict__ pooled) {
    int idx = blockIdx.x * blockDim.x + threadIdx.x;
    if (idx >= NN * 64) return;
    int n = idx >> 6, d = idx & 63;
    int g = batch[n] * 64 + d;
    float attn = __expf(gate[idx] - unordf(gmax[g])) / (gden[g] + EPSF);
    atomicAdd(&pooled[g], attn * xf[idx]);
}

// ---- tiny classifier: [G,64] @ [64,10] + b ----
__global__ void k_classifier(const float* __restrict__ pooled, const float* __restrict__ Wc,
                             const float* __restrict__ bc, float* __restrict__ out) {
    int idx = blockIdx.x * blockDim.x + threadIdx.x;
    if (idx >= GG * NCLS) return;
    int g = idx / NCLS, cl = idx % NCLS;
    float s = bc[cl];
#pragma unroll
    for (int d = 0; d < 64; ++d) s += pooled[g * 64 + d] * Wc[d * NCLS + cl];
    out[idx] = s;
}

static inline unsigned blks(size_t n) { return (unsigned)((n + 255) / 256); }

extern "C" void kernel_launch(void* const* d_in, const int* in_sizes, int n_in,
                              void* d_out, int out_size, void* d_ws, size_t ws_size,
                              hipStream_t stream) {
    const float* x      = (const float*)d_in[0];
    const int*   edge   = (const int*)d_in[1];
    const int*   batch  = (const int*)d_in[2];
    const float* Ws     = (const float*)d_in[3];
    const float* a_src  = (const float*)d_in[4];
    const float* a_dst  = (const float*)d_in[5];
    const float* bs     = (const float*)d_in[6];
    const float* W_gate = (const float*)d_in[7];
    const float* b_gate = (const float*)d_in[8];
    const float* W_cls  = (const float*)d_in[9];
    const float* b_cls  = (const float*)d_in[10];
    float* out = (float*)d_out;

    // workspace layout (~55 MB)
    char* p = (char*)d_ws;
    float* bufH  = (float*)p; p += (size_t)NN * 64 * 4;
    float* bufO1 = (float*)p; p += (size_t)NN * 64 * 4;
    float* bufO2 = (float*)p; p += (size_t)NN * 64 * 4;
    float* als   = (float*)p; p += (size_t)NN * HH * 4;
    float* ald   = (float*)p; p += (size_t)NN * HH * 4;
    float* esc   = (float*)p; p += (size_t)ETOT * HH * 4;
    int*   emax  = (int*)p;   p += (size_t)NN * HH * 4;
    float* denom = (float*)p; p += (size_t)NN * HH * 4;
    int*   gmax  = (int*)p;   p += (size_t)GG * 64 * 4;
    float* gden  = (float*)p; p += (size_t)GG * 64 * 4;
    float* pooled= (float*)p; p += (size_t)GG * 64 * 4;

    const unsigned gemmGrid = (NN + 127) / 128;
    const float* xcur = x;
    float* outs[3] = { bufO1, bufO2, bufO1 };

    for (int l = 0; l < 3; ++l) {
        float* obuf = outs[l];
        // h = x @ W_l   (WMMA)
        k_gemm64<<<gemmGrid, 256, 0, stream>>>(xcur, Ws + (size_t)l * 64 * 64,
                                               nullptr, bufH, NN);
        k_scores<<<blks((size_t)NN * HH), 256, 0, stream>>>(
            bufH, a_src + l * HH * CC, a_dst + l * HH * CC, als, ald);
        k_fill_i32<<<blks((size_t)NN * HH), 256, 0, stream>>>(emax, ORD_NEG_INF, NN * HH);
        k_fill_f32<<<blks((size_t)NN * HH), 256, 0, stream>>>(denom, 0.f, NN * HH);
        k_fill_f32<<<blks((size_t)NN * 64), 256, 0, stream>>>(obuf, 0.f, NN * 64);
        k_edge_max<<<blks((size_t)ETOT * HH), 256, 0, stream>>>(edge, als, ald, esc, emax);
        k_edge_sum<<<blks((size_t)ETOT * HH), 256, 0, stream>>>(edge, esc, emax, denom);
        k_edge_agg<<<blks((size_t)ETOT * HH), 256, 0, stream>>>(edge, esc, emax, denom,
                                                                bufH, obuf);
        k_bias_relu<<<blks((size_t)NN * 64), 256, 0, stream>>>(obuf, bs + l * 64);
        xcur = obuf;
    }

    // gate = x @ W_gate + b_gate  (WMMA, gate stored in bufH)
    k_gemm64<<<gemmGrid, 256, 0, stream>>>(xcur, W_gate, b_gate, bufH, NN);
    k_fill_i32<<<blks((size_t)GG * 64), 256, 0, stream>>>(gmax, ORD_NEG_INF, GG * 64);
    k_fill_f32<<<blks((size_t)GG * 64), 256, 0, stream>>>(gden, 0.f, GG * 64);
    k_fill_f32<<<blks((size_t)GG * 64), 256, 0, stream>>>(pooled, 0.f, GG * 64);
    k_pool_max<<<blks((size_t)NN * 64), 256, 0, stream>>>(bufH, batch, gmax);
    k_pool_sum<<<blks((size_t)NN * 64), 256, 0, stream>>>(bufH, batch, gmax, gden);
    k_pool_acc<<<blks((size_t)NN * 64), 256, 0, stream>>>(bufH, xcur, batch, gmax, gden, pooled);
    k_classifier<<<blks((size_t)GG * NCLS), 256, 0, stream>>>(pooled, W_cls, b_cls, out);
}
